// RNN_61770219651190
// MI455X (gfx1250) — matrix operand order repaired and checked by reading, not compiled, not run on hardware
//
#include <hip/hip_runtime.h>
#include <hip/hip_bf16.h>

typedef __attribute__((ext_vector_type(16))) _Float16 v16h;
typedef __attribute__((ext_vector_type(8)))  _Float16 v8h;
typedef __attribute__((ext_vector_type(8)))  float    v8f;

// Problem constants (from reference)
#define T_TOT   300
#define FIXT    50
#define SAMP    50
#define DELAY   100
#define TESTT   50
#define RESP    50
#define M_MODELS 256
#define N_CELL   128

// LDS row stride (halves) for the f16 state mirror: 136*2=272B rows -> 16B
// aligned rows, and conflict-free banks across the 16 B-columns for b128 loads.
#define R16_STRIDE 136

__global__ __launch_bounds__(256)
void rnn_scan_kernel(const float* __restrict__ x,          // (4,2,2)
                     const float* __restrict__ noise,      // (300,256,128)
                     const float* __restrict__ Wrec,       // (256,128,128)
                     const float* __restrict__ brec,       // (256,128)
                     const float* __restrict__ Winp,       // (256,128,2)
                     const float* __restrict__ Wout,       // (256,2,128)
                     const float* __restrict__ Wmem,       // (256,2,128)
                     float* __restrict__ out)              // rm(100,4,256,2) ++ rs(50,4,256,2)
{
    __shared__ _Float16 r16[16 * R16_STRIDE]; // f16 state mirror, cols 4..15 stay zero
    __shared__ float    r32[4 * N_CELL];      // f32 state for projections
    __shared__ float    nz[N_CELL];           // per-step noise stage

    const int model = blockIdx.x;
    const int tid   = threadIdx.x;
    const int wave  = tid >> 5;
    const int lane  = tid & 31;
    const int half  = lane >> 4;    // K-half selector for A/B fragments
    const int n     = lane & 15;    // B/C/D column (batch index when < 4)
    const int mbase = 16 * wave + 8 * half; // first C/D row this lane owns

    // ---- zero-init LDS state ----
    for (int i = tid; i < 16 * R16_STRIDE; i += 256) r16[i] = (_Float16)0.0f;
    for (int i = tid; i < 4 * N_CELL;      i += 256) r32[i] = 0.0f;

    // ---- Load recurrent weights once into persistent WMMA A-fragments ----
    // A layout (16-bit, 16x32): M = lane&15 for both lane halves;
    // half h of v16h -> K = (h&7) + ((h>>3)<<4) + 8*(lane>>4), per K-chunk kc.
    v16h afrag[4];
    {
        const int    mA   = 16 * wave + n;
        const float* wrow = Wrec + (size_t)model * (N_CELL * N_CELL) + (size_t)mA * N_CELL;
#pragma unroll
        for (int kc = 0; kc < 4; ++kc) {
#pragma unroll
            for (int h = 0; h < 16; ++h) {
                const int k = kc * 32 + (h & 7) + ((h >> 3) << 4) + (half << 3);
                afrag[kc][h] = (_Float16)wrow[k];
            }
        }
    }

    // ---- Per-lane constants: bias, stimulus drives, state registers ----
    float bias_v[8], ds_v[8], dt_v[8], r_reg[8];
#pragma unroll
    for (int v = 0; v < 8; ++v) {
        const int m = mbase + v;
        bias_v[v] = brec[(size_t)model * N_CELL + m];
        r_reg[v]  = 0.0f;
        ds_v[v]   = 0.0f;
        dt_v[v]   = 0.0f;
        if (n < 4) {
            const float wi0 = Winp[(size_t)model * 256 + m * 2 + 0];
            const float wi1 = Winp[(size_t)model * 256 + m * 2 + 1];
            // x[b, phase, s] flat = b*4 + phase*2 + s
            ds_v[v] = wi0 * x[n * 4 + 0] + wi1 * x[n * 4 + 1]; // sample drive
            dt_v[v] = wi0 * x[n * 4 + 2] + wi1 * x[n * 4 + 3]; // test drive
        }
    }

    // ---- Projection weights: 8 waves <-> 8 (o, batch) pairs ----
    const int o     = wave & 1;
    const int bproj = wave >> 1;
    float wm_r[4], wo_r[4];
#pragma unroll
    for (int j = 0; j < 4; ++j) {
        wm_r[j] = Wmem[(size_t)model * 256 + o * N_CELL + lane + 32 * j];
        wo_r[j] = Wout[(size_t)model * 256 + o * N_CELL + lane + 32 * j];
    }

    __syncthreads();

    // Per-lane noise source/destination (waves 0..3 stage one cell each).
    const float* noise_src = noise + (size_t)model * N_CELL + tid;
    const unsigned nz_lds  = (unsigned)(unsigned long long)&nz[tid & (N_CELL - 1)];

    for (int t = 0; t < T_TOT; ++t) {
        // ---- Issue async global->LDS noise copy first (ASYNCcnt path) ----
        if (tid < N_CELL) {
            asm volatile("global_load_async_to_lds_b32 %0, %1, off"
                         :: "v"(nz_lds), "v"(noise_src) : "memory");
            if (t + 1 < T_TOT)
                __builtin_prefetch(noise_src + (size_t)(M_MODELS * N_CELL), 0, 1);
            noise_src += (size_t)(M_MODELS * N_CELL);
        }

        // ---- Build B fragments from f16 LDS state ----
        // B layout (16-bit, 32x16): N = lane&15, K = (h&7) + ((h>>3)<<4) + 8*(lane>>4), chunk kc.
        v16h bfrag[4];
#pragma unroll
        for (int kc = 0; kc < 4; ++kc) {
            const _Float16* p  = &r16[n * R16_STRIDE + kc * 32 + (half << 3)];
            const v8h lo = *(const v8h*)(p);
            const v8h hi = *(const v8h*)(p + 16);
#pragma unroll
            for (int i = 0; i < 8; ++i) { bfrag[kc][i] = lo[i]; bfrag[kc][i + 8] = hi[i]; }
        }

        // Async noise must land in LDS before the barrier releases readers.
        if (tid < N_CELL)   // wave-uniform guard (waves 0..3)
            asm volatile("s_wait_asynccnt 0" ::: "memory");
        __syncthreads();   // B reads done before state overwrite; noise staged

        // ---- y(16x16) = W_tile(16x128) @ R(128x16), two independent
        //      accumulator chains (depth 2) to cut matrix-pipe RAW stalls ----
        v8f acc0 = {}, acc1 = {};
        acc0 = __builtin_amdgcn_wmma_f32_16x16x32_f16(false, afrag[0], false, bfrag[0],
                                                      (short)0, acc0, false, false);
        acc1 = __builtin_amdgcn_wmma_f32_16x16x32_f16(false, afrag[1], false, bfrag[1],
                                                      (short)0, acc1, false, false);
        acc0 = __builtin_amdgcn_wmma_f32_16x16x32_f16(false, afrag[2], false, bfrag[2],
                                                      (short)0, acc0, false, false);
        acc1 = __builtin_amdgcn_wmma_f32_16x16x32_f16(false, afrag[3], false, bfrag[3],
                                                      (short)0, acc1, false, false);
        const v8f acc = acc0 + acc1;

        // ---- Pointwise Euler update: r' = 0.8 r + 0.2 relu(y+b+drive) + 0.02 nz ----
        const float ms = (t >= FIXT && t < FIXT + SAMP) ? 1.0f : 0.0f;
        const float mt = (t >= FIXT + SAMP + DELAY && t < FIXT + SAMP + DELAY + TESTT) ? 1.0f : 0.0f;
        if (n < 4) {
#pragma unroll
            for (int v = 0; v < 8; ++v) {
                const int m = mbase + v;
                float y = acc[v] + bias_v[v] + ms * ds_v[v] + mt * dt_v[v];
                y = fmaxf(y, 0.0f);
                const float rn = 0.8f * r_reg[v] + 0.2f * y + 0.02f * nz[m];
                r_reg[v] = rn;
                r16[n * R16_STRIDE + m] = (_Float16)rn;
                r32[n * N_CELL + m]     = rn;
            }
        }
        __syncthreads();   // r_{t+1} visible to all waves

        // ---- On-the-fly output projections in the delay / response windows ----
        const bool dly = (t >= FIXT + SAMP) && (t < FIXT + SAMP + DELAY);
        const bool rsp = (t >= T_TOT - RESP);
        if (dly || rsp) {
            float s = 0.0f;
#pragma unroll
            for (int j = 0; j < 4; ++j) {
                const float w = dly ? wm_r[j] : wo_r[j];
                s += w * r32[bproj * N_CELL + lane + 32 * j];
            }
#pragma unroll
            for (int off = 16; off > 0; off >>= 1)
                s += __shfl_xor(s, off, 32);
            if (lane == 0) {
                size_t idx;
                if (dly)  idx = ((size_t)(t - (FIXT + SAMP)) * 4 + bproj) * (M_MODELS * 2)
                              + model * 2 + o;
                else      idx = (size_t)DELAY * 4 * M_MODELS * 2
                              + ((size_t)(t - (T_TOT - RESP)) * 4 + bproj) * (M_MODELS * 2)
                              + model * 2 + o;
                out[idx] = s;
            }
        }
    }
}

extern "C" void kernel_launch(void* const* d_in, const int* in_sizes, int n_in,
                              void* d_out, int out_size, void* d_ws, size_t ws_size,
                              hipStream_t stream) {
    const float* x     = (const float*)d_in[0];
    const float* noise = (const float*)d_in[1];
    const float* Wrec  = (const float*)d_in[2];
    const float* brec  = (const float*)d_in[3];
    const float* Winp  = (const float*)d_in[4];
    const float* Wout  = (const float*)d_in[5];
    const float* Wmem  = (const float*)d_in[6];
    float* out = (float*)d_out;

    rnn_scan_kernel<<<M_MODELS, 256, 0, stream>>>(x, noise, Wrec, brec, Winp, Wout, Wmem, out);
}
